// MAC_59966333387032
// MI455X (gfx1250) — compile-verified, matrix operand order checked by reading, not run on hardware
//
#include <hip/hip_runtime.h>
#include <hip/hip_bf16.h>

// Problem constants (match reference)
#define BSZ   256      // batch
#define IDIM  16384    // flattened input size
#define CMS   32       // num CMs
#define NN    64       // neurons per CM
#define KSPLIT 8
#define KCH   (IDIM / KSPLIT)   // 2048 K per split
#define KB    64                // K per pipeline stage (= two WMMA K-steps, 16 KB fp32)
#define DEPTH 2                 // async prefetch distance (stages in flight)
#define NBUF  3                 // LDS ring buffers (DEPTH+1), 3 x 16 KB = 48 KB

typedef __attribute__((ext_vector_type(16))) __bf16 v16bf;
typedef __attribute__((ext_vector_type(8)))  __bf16 v8bf;
typedef __attribute__((ext_vector_type(8)))  float  v8f;

// Toggle: 1 = gfx1250 async global->LDS copies (ASYNCcnt pipeline),
//         0 = portable fallback (register load + ds_store)
#define USE_ASYNC_LDS 1

#if defined(__has_builtin)
#  if __has_builtin(__builtin_amdgcn_sched_barrier)
#    define SCHED_FENCE() __builtin_amdgcn_sched_barrier(0)
#  endif
#endif
#ifndef SCHED_FENCE
#  define SCHED_FENCE() asm volatile("" ::: "memory")
#endif

__device__ __forceinline__ __bf16 f32_to_bf16(float f) {
    union { float f; unsigned u; } a; a.f = f;
    unsigned r = a.u + 0x7FFFu + ((a.u >> 16) & 1u);   // round-to-nearest-even
    union { unsigned short s; __bf16 b; } o;
    o.s = (unsigned short)(r >> 16);
    return o.b;
}

// ---------------------------------------------------------------------------
// Kernel 0: per-row sum of x -> inv_sum[b] = 1/sum (0 if sum==0, nan_to_num)
// ---------------------------------------------------------------------------
__global__ __launch_bounds__(256)
void rowsum_kernel(const float* __restrict__ x, float* __restrict__ inv_sum) {
    const int b = blockIdx.x;
    const float* row = x + (size_t)b * IDIM;
    float s = 0.f;
    for (int i = threadIdx.x; i < IDIM; i += 256) s += row[i];
    __shared__ float red[256];
    red[threadIdx.x] = s;
    __syncthreads();
    for (int off = 128; off > 0; off >>= 1) {
        if (threadIdx.x < off) red[threadIdx.x] += red[threadIdx.x + off];
        __syncthreads();
    }
    if (threadIdx.x == 0) {
        float t = red[0];
        inv_sum[b] = (t != 0.f) ? (1.f / t) : 0.f;
    }
}

// ---------------------------------------------------------------------------
// Kernel 1: x_bf16[b,i] = bf16(x[b,i] * inv_sum[b])   (float4 in, 4x bf16 out)
// ---------------------------------------------------------------------------
__global__ __launch_bounds__(256)
void convert_kernel(const float* __restrict__ x, const float* __restrict__ inv_sum,
                    __bf16* __restrict__ xb) {
    const size_t idx4 = ((size_t)blockIdx.x * 256 + threadIdx.x) * 4;
    const int b = (int)(idx4 >> 14);                 // IDIM = 2^14, float4 never crosses rows
    const float sc = inv_sum[b];
    const float4 v = *(const float4*)(x + idx4);
    union { unsigned short h[4]; unsigned long long u; } pk;
    union { __bf16 b; unsigned short s; } c0, c1, c2, c3;
    c0.b = f32_to_bf16(v.x * sc);
    c1.b = f32_to_bf16(v.y * sc);
    c2.b = f32_to_bf16(v.z * sc);
    c3.b = f32_to_bf16(v.w * sc);
    pk.h[0] = c0.s; pk.h[1] = c1.s; pk.h[2] = c2.s; pk.h[3] = c3.s;
    *(unsigned long long*)(xb + idx4) = pk.u;        // 8B aligned (idx4*2)
}

// ---------------------------------------------------------------------------
// Stage copy: one contiguous 16 KB weight chunk (KB x NN fp32, row-major (k,n))
// from global into LDS. 256 threads x 64 B each (4 async b128 copies per lane).
// ---------------------------------------------------------------------------
__device__ __forceinline__ void issue_stage(const float* __restrict__ Wc, int k0,
                                            float* lds_dst, int tid) {
#if USE_ASYNC_LDS
    // gfx1250 async copy, tracked by ASYNCcnt (CDNA5 ISA).
    // Generic shared pointer: low 32 bits are the wave-relative LDS byte address.
    unsigned lds0 = (unsigned)(uintptr_t)lds_dst + (unsigned)(tid * 64);
    unsigned long long g0 =
        (unsigned long long)(uintptr_t)(Wc + (size_t)k0 * NN) + (unsigned long long)(tid * 64);
    asm volatile("global_load_async_to_lds_b128 %0, %1, off"            :: "v"(lds0), "v"(g0) : "memory");
    asm volatile("global_load_async_to_lds_b128 %0, %1, off offset:16"  :: "v"(lds0), "v"(g0) : "memory");
    asm volatile("global_load_async_to_lds_b128 %0, %1, off offset:32"  :: "v"(lds0), "v"(g0) : "memory");
    asm volatile("global_load_async_to_lds_b128 %0, %1, off offset:48"  :: "v"(lds0), "v"(g0) : "memory");
#else
    const float4* s4 = (const float4*)(Wc + (size_t)k0 * NN) + tid * 4;
    float4* d4 = (float4*)lds_dst + tid * 4;
    d4[0] = s4[0]; d4[1] = s4[1]; d4[2] = s4[2]; d4[3] = s4[3];
#endif
}

// Wait until this wave's oldest in-flight stage (4 async ops) has landed.
__device__ __forceinline__ void wait_oldest_stage() {
#if USE_ASYNC_LDS
    asm volatile("s_wait_asynccnt 4" ::: "memory");   // 4*(DEPTH-1) ops may remain
#endif
}

// ---------------------------------------------------------------------------
// A-fragment loader: 8 M-tiles, 2 x b128 per tile from L2-resident x_bf16.
// Lanes 0-15 carry M=lane with K in {0..7,16..23}; lanes 16-31 K in {8..15,24..31}
// (xrow already folds in the hi*8 lane offset).
// ---------------------------------------------------------------------------
__device__ __forceinline__ void loadA(v16bf (&a)[8], const __bf16* __restrict__ xrow, int k0) {
#pragma unroll
    for (int t = 0; t < 8; ++t) {
        const __bf16* xr = xrow + (size_t)t * 16 * IDIM + k0;
        v8bf alo = *(const v8bf*)(xr);          // K offsets {0..7} (or {8..15})
        v8bf ahi = *(const v8bf*)(xr + 16);     // K offsets {16..23} (or {24..31})
#pragma unroll
        for (int j = 0; j < 8; ++j) { a[t][j] = alo[j]; a[t][8 + j] = ahi[j]; }
    }
}

// B fragment from LDS chunk (row-major KBxNN fp32), fp32->bf16 in-register.
__device__ __forceinline__ v16bf buildB(const float* __restrict__ Bs, int hi, int n) {
    v16bf b;
#pragma unroll
    for (int j = 0; j < 8; ++j) {
        b[j]     = f32_to_bf16(Bs[(hi * 8 + j)      * NN + n]);
        b[8 + j] = f32_to_bf16(Bs[(hi * 8 + 16 + j) * NN + n]);
    }
    return b;
}

// ---------------------------------------------------------------------------
// Kernel 2: split-K GEMM via v_wmma_f32_16x16x32_bf16.
//   grid = (C, KSPLIT); block = 256 threads (8 waves).
//   Block (c,s): all 256 batch rows x 64 cols of CM c, K = [s*2048, (s+1)*2048).
//   Wave w: rows [ (w>>2)*128 , +128 ), cols [ (w&3)*16 , +16 )  -> 8 C-tiles.
//   Weights stream HBM exactly once via a depth-2 ring of async->LDS copies;
//   A fragments are software-pipelined (aX/aY rotation) so WMMAs overlap both
//   the HBM (ASYNCcnt) and L2 (LOADcnt) traffic.
// ---------------------------------------------------------------------------
__global__ __launch_bounds__(256)
void gemm_kernel(const float* __restrict__ W, const __bf16* __restrict__ Xb,
                 float* __restrict__ partial) {
    const int c    = blockIdx.x;
    const int s    = blockIdx.y;
    const int tid  = threadIdx.x;
    const int wave = tid >> 5;
    const int lane = tid & 31;
    const int lanelo = lane & 15;
    const int hi     = lane >> 4;            // 0: lanes 0-15, 1: lanes 16-31
    const int ncol0  = (wave & 3) * 16;
    const int mbase  = (wave >> 2) * 128;

    __shared__ float lds_w[NBUF][KB * NN];   // 3 x 16 KB ring

    v8f acc[8] = {};                         // 8 MxN tiles of 16x16 f32

    const float* Wc = W + (size_t)c * IDIM * NN;
    const int n     = ncol0 + lanelo;        // this lane's column (B-frag / D-tile)
    const int kbeg  = s * KCH;
    const int nstages = KCH / KB;            // 32 stages of 64 K

    // Prologue: stages 0..DEPTH-1 in flight, first A K-step in registers.
#pragma unroll
    for (int p = 0; p < DEPTH; ++p)
        issue_stage(Wc, kbeg + p * KB, lds_w[p], tid);

    const __bf16* xrow = Xb + (size_t)(mbase + lanelo) * IDIM + hi * 8;
    v16bf aX[8], aY[8];
    loadA(aX, xrow, kbeg);

    for (int si = 0; si < nstages; ++si) {
        const float* Bs = lds_w[si % NBUF];

        // A prefetch for this stage's 2nd K-step: issue before the barrier so
        // the loads run underneath it.
        loadA(aY, xrow, kbeg + si * KB + 32);

        // Oldest async stage (si) landed for this wave; barrier makes all
        // waves' copies visible AND proves stage si-1's buffer is consumed,
        // so re-issuing into (si+DEPTH)%NBUF is safe. Tail re-issues a clamped
        // dummy stage to keep the in-flight count uniform (drained by endpgm).
        wait_oldest_stage();
        __syncthreads();
        const int sn = (si + DEPTH < nstages) ? (si + DEPTH) : (nstages - 1);
        issue_stage(Wc, kbeg + sn * KB, lds_w[(si + DEPTH) % NBUF], tid);

        // ---- K-step 0 of this stage: consume aX
        {
            v16bf bfrag = buildB(Bs, hi, n);
            SCHED_FENCE();                       // keep prefetch loads above WMMAs
#pragma unroll
            for (int t = 0; t < 8; ++t)
                acc[t] = __builtin_amdgcn_wmma_f32_16x16x32_bf16(
                             false, aX[t], false, bfrag, (short)0, acc[t], false, false);
        }

        // A prefetch for next stage's 1st K-step (clamped reload at tail).
        const int kx = (si + 1 < nstages) ? (si + 1) * KB : si * KB;
        loadA(aX, xrow, kbeg + kx);

        // ---- K-step 1 of this stage: consume aY
        {
            v16bf bfrag = buildB(Bs + 32 * NN, hi, n);
            SCHED_FENCE();
#pragma unroll
            for (int t = 0; t < 8; ++t)
                acc[t] = __builtin_amdgcn_wmma_f32_16x16x32_bf16(
                             false, aY[t], false, bfrag, (short)0, acc[t], false, false);
        }
    }

    // ---- store partials: partial[s][m][c][n]
    // C/D layout: VGPR r holds M = tile_row + r + 8*hi, lane column = ncol0 + lanelo.
#pragma unroll
    for (int t = 0; t < 8; ++t) {
#pragma unroll
        for (int r = 0; r < 8; ++r) {
            const int m = mbase + t * 16 + r + hi * 8;
            partial[(((size_t)s * BSZ + m) * CMS + c) * NN + n] = acc[t][r];
        }
    }
}

// ---------------------------------------------------------------------------
// Kernel 3: sum K-split partials, argmax over n, write one-hot output.
//   grid = B*C blocks, 64 threads (one per neuron).
// ---------------------------------------------------------------------------
__global__ __launch_bounds__(64)
void reduce_argmax_kernel(const float* __restrict__ partial, float* __restrict__ out) {
    const int bc = blockIdx.x;
    const int b  = bc / CMS;
    const int c  = bc % CMS;
    const int nidx = threadIdx.x;

    float v = 0.f;
#pragma unroll
    for (int s = 0; s < KSPLIT; ++s)
        v += partial[(((size_t)s * BSZ + b) * CMS + c) * NN + nidx];

    __shared__ float sv[64];
    __shared__ int   si[64];
    sv[nidx] = v; si[nidx] = nidx;
    __syncthreads();
    for (int off = 32; off > 0; off >>= 1) {
        if (nidx < off) {
            // keep lower index on ties (matches argmax first-max semantics)
            if (sv[nidx + off] > sv[nidx]) { sv[nidx] = sv[nidx + off]; si[nidx] = si[nidx + off]; }
        }
        __syncthreads();
    }
    const int amax = si[0];
    out[(size_t)bc * NN + nidx] = (nidx == amax) ? 1.f : 0.f;
}

// ---------------------------------------------------------------------------
extern "C" void kernel_launch(void* const* d_in, const int* in_sizes, int n_in,
                              void* d_out, int out_size, void* d_ws, size_t ws_size,
                              hipStream_t stream) {
    const float* x = (const float*)d_in[0];   // [B, M, CIN, NIN] = [B, I] fp32
    const float* w = (const float*)d_in[1];   // [C, I, N] fp32
    float* out = (float*)d_out;               // [B, C, N] fp32 one-hot

    // scratch layout: inv_sum (1 KB, padded to 4 KB) | x_bf16 (8 MB) | partial (16 MB)
    char* ws = (char*)d_ws;
    float*  inv_sum = (float*)ws;
    __bf16* xb      = (__bf16*)(ws + 4096);
    float*  partial = (float*)(ws + 4096 + (size_t)BSZ * IDIM * sizeof(__bf16));

    rowsum_kernel<<<BSZ, 256, 0, stream>>>(x, inv_sum);

    const int total4 = (BSZ * IDIM) / 4;            // 1,048,576 float4s
    convert_kernel<<<total4 / 256, 256, 0, stream>>>(x, inv_sum, xb);

    gemm_kernel<<<dim3(CMS, KSPLIT), 256, 0, stream>>>(w, xb, partial);

    reduce_argmax_kernel<<<BSZ * CMS, 64, 0, stream>>>(partial, out);
}
